// SlidingWindowCausalAttention_42649025249825
// MI455X (gfx1250) — matrix operand order repaired and checked by reading, not compiled, not run on hardware
//
#include <hip/hip_runtime.h>
#include <hip/hip_bf16.h>
#include <math.h>

typedef __bf16 bf16_t;
typedef __attribute__((ext_vector_type(16))) __bf16 v16bf;
typedef __attribute__((ext_vector_type(8)))  __bf16 v8bf;
typedef __attribute__((ext_vector_type(8)))  float  v8f;
typedef __attribute__((ext_vector_type(4)))  int    v4i;

#define WMMA_BF16(a, b, c) \
  __builtin_amdgcn_wmma_f32_16x16x32_bf16(false, (a), false, (b), (short)0, (c), false, false)

// --- CDNA5 async global->LDS path (probe-confirmed builtin; safe fallback) ---
#if defined(__has_builtin)
#if __has_builtin(__builtin_amdgcn_global_load_async_to_lds_b128)
#define ASYNC_LDS 1
#endif
#endif
#ifndef ASYNC_LDS
#define ASYNC_LDS 0
#endif

typedef __attribute__((address_space(1))) v4i* gv4ip;   // global v4i*
typedef __attribute__((address_space(3))) v4i* lv4ip;   // LDS v4i*

static constexpr int Bb = 2, Ss = 2048, Ee = 1024, Hh = 16, Dd = 64;
static constexpr int Mm = Bb * Ss;      // 4096 rows
static constexpr int NF = 3 * Ee;       // 3072 qkv outputs
static constexpr int SB_STRIDE = 40;    // 32 cols + 8 pad (80B rows, 16B aligned)
static constexpr int SB_TILE   = 64 * SB_STRIDE;   // one buffer, bf16 elems

__device__ __forceinline__ v16bf cat8(v8bf lo, v8bf hi) {
  return __builtin_shufflevector(lo, hi, 0,1,2,3,4,5,6,7,8,9,10,11,12,13,14,15);
}

// ---------------------------------------------------------------- fp32 -> bf16
__global__ void cvt_f32_bf16(const float* __restrict__ in, bf16_t* __restrict__ out, int n) {
  int i = blockIdx.x * blockDim.x + threadIdx.x;
  int stride = gridDim.x * blockDim.x;
  for (; i < n; i += stride) out[i] = (bf16_t)in[i];
}

// Stage one 64(f) x 32(e) bf16 weight tile into LDS (128 threads, 32B each,
// 2 async b128 transfers per thread => asynccnt +2 per wave).
__device__ __forceinline__ void stage_B(const bf16_t* __restrict__ Bw,
                                        bf16_t* __restrict__ sB, int f0, int e0) {
  const int t = threadIdx.x;            // 0..127
  const int r = t >> 1, half = (t & 1) * 16;
  const bf16_t* src = Bw + (size_t)(f0 + r) * Ee + e0 + half;
  bf16_t* dst = sB + r * SB_STRIDE + half;
#if ASYNC_LDS
  gv4ip g = (gv4ip)src;
  lv4ip l = (lv4ip)dst;
  __builtin_amdgcn_global_load_async_to_lds_b128(g,     l,     0, 0);
  __builtin_amdgcn_global_load_async_to_lds_b128(g + 1, l + 1, 0, 0);
#else
  *(v16bf*)dst = *(const v16bf*)src;
#endif
}

// Shared 32(m) x 64(f) wave-tile GEMM core: C = A(m,e) * B(f,e)^T, bf16 in, f32 acc.
// 4 waves/block share the B tile via double-buffered async LDS staging; the next
// tile's copies stay in flight (s_wait_asynccnt 2) while WMMAs consume the
// current one. Named accumulators keep the allocator from shuffling WMMA
// destinations (avoids WMMA->VALU hazard NOPs in the hot loop).
__device__ __forceinline__ void gemm_core_32x64(const bf16_t* __restrict__ A,
                                                const bf16_t* __restrict__ Bw,
                                                bf16_t* __restrict__ sB,   // 2 tiles
                                                int m0, int f0,
                                                v8f& c00, v8f& c01, v8f& c02, v8f& c03,
                                                v8f& c10, v8f& c11, v8f& c12, v8f& c13) {
  const int lane = threadIdx.x & 31;
  const bool hiHalf = lane >= 16;
  const int qLane = lane & 15;
  const int loA = hiHalf ? 16 : 0;
  const int loB = hiHalf ? 16 : 0;
  const bf16_t* arow0 = A + (size_t)(m0 + qLane) * Ee;
  const bf16_t* arow1 = arow0 + (size_t)16 * Ee;
  const bf16_t* brow = sB + (size_t)qLane * SB_STRIDE + loB;

  constexpr int NIT = Ee / 32;          // 32 K-steps
  stage_B(Bw, sB, f0, 0);               // prologue: fill buffer 0
  for (int i = 0; i < NIT; ++i) {
    const int e0 = i * 32;
    const bf16_t* cur = brow + (i & 1) * SB_TILE;
    bf16_t* nxt = sB + ((i + 1) & 1) * SB_TILE;
    // all waves finished reading nxt's old contents (iteration i-1)
    __syncthreads();
    if (i + 1 < NIT) {
      stage_B(Bw, nxt, f0, e0 + 32);    // overlap next copy with this compute
#if ASYNC_LDS
      __builtin_amdgcn_s_wait_asynccnt(2);   // cur's 2 done; nxt's may remain
#endif
    } else {
#if ASYNC_LDS
      __builtin_amdgcn_s_wait_asynccnt(0);
#endif
    }
    __syncthreads();                    // cur fully landed for every wave

    v16bf a0 = cat8(*(const v8bf*)(arow0 + e0 + loA), *(const v8bf*)(arow0 + e0 + 8 + loA));
    v16bf a1 = cat8(*(const v8bf*)(arow1 + e0 + loA), *(const v8bf*)(arow1 + e0 + 8 + loA));
    v16bf b0 = *(const v16bf*)(cur + 0  * 16 * SB_STRIDE);
    v16bf b1 = *(const v16bf*)(cur + 1  * 16 * SB_STRIDE);
    v16bf b2 = *(const v16bf*)(cur + 2  * 16 * SB_STRIDE);
    v16bf b3 = *(const v16bf*)(cur + 3  * 16 * SB_STRIDE);
    c00 = WMMA_BF16(a0, b0, c00);
    c10 = WMMA_BF16(a1, b0, c10);
    c01 = WMMA_BF16(a0, b1, c01);
    c11 = WMMA_BF16(a1, b1, c11);
    c02 = WMMA_BF16(a0, b2, c02);
    c12 = WMMA_BF16(a1, b2, c12);
    c03 = WMMA_BF16(a0, b3, c03);
    c13 = WMMA_BF16(a1, b3, c13);
  }
}

// ------------------------------------------------- QKV projection (NT GEMM)
// Routes t=0 -> Q[b,h,s,d], t=1 -> K[b,h,s,d], t=2 -> Vt[b,h,d,s] (all bf16).
__global__ __launch_bounds__(128) void qkv_gemm(const bf16_t* __restrict__ xb,
                                                const bf16_t* __restrict__ wqb,
                                                bf16_t* __restrict__ Qb,
                                                bf16_t* __restrict__ Kb,
                                                bf16_t* __restrict__ Vt) {
  __shared__ bf16_t sB[2 * SB_TILE];
  const int wave = threadIdx.x >> 5;
  const int mblock = blockIdx.x & 31;   // 32 blocks of 128 rows
  const int nblock = blockIdx.x >> 5;   // 48 tiles of 64 cols
  const int m0 = mblock * 128 + wave * 32;
  const int f0 = nblock * 64;
  v8f c00 = {}, c01 = {}, c02 = {}, c03 = {};
  v8f c10 = {}, c11 = {}, c12 = {}, c13 = {};
  gemm_core_32x64(xb, wqb, sB, m0, f0, c00, c01, c02, c03, c10, c11, c12, c13);

  const int lane = threadIdx.x & 31;
  const bool hiHalf = lane >= 16;
  const int qLane = lane & 15;
  const int t  = f0 >> 10;
  const int h  = (f0 >> 6) & 15;
  const int bi = m0 >> 11;              // batch (blocks never straddle batches)
  const v8f cc[2][4] = {{c00, c01, c02, c03}, {c10, c11, c12, c13}};
  for (int mh = 0; mh < 2; ++mh) {
    for (int nt = 0; nt < 4; ++nt) {
      const int d = nt * 16 + qLane;
      for (int g = 0; g < 8; ++g) {
        const int s = (m0 + mh * 16 + g + (hiHalf ? 8 : 0)) & (Ss - 1);
        const bf16_t val = (bf16_t)cc[mh][nt][g];
        if (t == 0)      Qb[(((size_t)bi * Hh + h) * Ss + s) * Dd + d] = val;
        else if (t == 1) Kb[(((size_t)bi * Hh + h) * Ss + s) * Dd + d] = val;
        else             Vt[(((size_t)bi * Hh + h) * Dd + d) * Ss + s] = val;
      }
    }
  }
}

// -------------------------------------------------------- causal flash attention
// One wave per (b, h, 16-query tile). Scores computed transposed (T = K * Q^T)
// so the C-layout of T is already the A-fragment layout for P*V.
__global__ __launch_bounds__(128) void flash_attn(const bf16_t* __restrict__ Qb,
                                                  const bf16_t* __restrict__ Kb,
                                                  const bf16_t* __restrict__ Vt,
                                                  bf16_t* __restrict__ ab) {
  const int lane = threadIdx.x & 31;
  const int wid  = blockIdx.x * (blockDim.x >> 5) + (threadIdx.x >> 5); // 0..4095
  const int it = wid & 127;
  const int h  = (wid >> 7) & 15;
  const int b  = wid >> 11;
  const int q0 = it * 16;
  const bool hiHalf = lane >= 16;
  const int qLane = lane & 15;
  const size_t bhQ = ((size_t)b * Hh + h) * (size_t)Ss * Dd;
  const size_t bhV = ((size_t)b * Hh + h) * (size_t)Dd * Ss;

  v16bf bq0, bq1;
  {
    const bf16_t* qrow = Qb + bhQ + (size_t)(q0 + qLane) * Dd;
    const int db = hiHalf ? 16 : 0;
    bq0 = *(const v16bf*)(qrow + db);        // d 0..31 slice
    bq1 = *(const v16bf*)(qrow + 32 + db);   // d 32..63 slice
  }

  v8f o0 = {}, o1 = {}, o2 = {}, o3 = {};
  float m_run = -1e30f, l_run = 0.f;
  const float scale = 0.125f;                 // 1/sqrt(64)
  const int nblocks = (q0 + 16 + 31) >> 5;    // keys up to q0+15, 32 keys/block

  for (int jb = 0; jb < nblocks; ++jb) {
    const int k0 = jb * 32;
    v8f t0 = {}, t1 = {};
    {
      const int loA = hiHalf ? 16 : 0;
      const bf16_t* krow0 = Kb + bhQ + (size_t)(k0 + qLane) * Dd;
      const bf16_t* krow1 = krow0 + (size_t)16 * Dd;
      for (int p = 0; p < 2; ++p) {
        const int dbl = p * 32 + loA;
        v16bf a0 = cat8(*(const v8bf*)(krow0 + dbl), *(const v8bf*)(krow0 + dbl + 8));
        v16bf a1 = cat8(*(const v8bf*)(krow1 + dbl), *(const v8bf*)(krow1 + dbl + 8));
        v16bf bq = (p == 0) ? bq0 : bq1;
        t0 = WMMA_BF16(a0, bq, t0);
        t1 = WMMA_BF16(a1, bq, t1);
      }
    }
    const int q = q0 + qLane;
    const int kbase = k0 + (hiHalf ? 8 : 0);
    float s0[8], s1[8];
    float lmax = -1e30f;
    for (int g = 0; g < 8; ++g) {
      const int kk = kbase + g;
      s0[g] = (kk      <= q) ? t0[g] * scale : -1e30f;
      s1[g] = (kk + 16 <= q) ? t1[g] * scale : -1e30f;
      lmax = fmaxf(lmax, fmaxf(s0[g], s1[g]));
    }
    lmax = fmaxf(lmax, __shfl_xor(lmax, 16));
    const float m_new = fmaxf(m_run, lmax);
    const float alpha = __expf(m_run - m_new);
    float psum = 0.f;
    v16bf pa;
    for (int g = 0; g < 8; ++g) {
      const float p0 = __expf(s0[g] - m_new);
      const float p1 = __expf(s1[g] - m_new);
      psum += p0 + p1;
      pa[g]     = (bf16_t)p0;
      pa[g + 8] = (bf16_t)p1;
    }
    psum += __shfl_xor(psum, 16);
    l_run = l_run * alpha + psum;
    m_run = m_new;
    for (int g = 0; g < 8; ++g) {
      const float ag = __shfl(alpha, g + (hiHalf ? 8 : 0));
      o0[g] *= ag; o1[g] *= ag; o2[g] *= ag; o3[g] *= ag;
    }
    const bf16_t* vp = Vt + bhV + (size_t)qLane * Ss + k0 + (hiHalf ? 8 : 0);
    o0 = WMMA_BF16(pa, cat8(*(const v8bf*)(vp + (size_t)0  * Ss), *(const v8bf*)(vp + (size_t)0  * Ss + 16)), o0);
    o1 = WMMA_BF16(pa, cat8(*(const v8bf*)(vp + (size_t)16 * Ss), *(const v8bf*)(vp + (size_t)16 * Ss + 16)), o1);
    o2 = WMMA_BF16(pa, cat8(*(const v8bf*)(vp + (size_t)32 * Ss), *(const v8bf*)(vp + (size_t)32 * Ss + 16)), o2);
    o3 = WMMA_BF16(pa, cat8(*(const v8bf*)(vp + (size_t)48 * Ss), *(const v8bf*)(vp + (size_t)48 * Ss + 16)), o3);
  }

  const float il = 1.f / l_run;
  for (int g = 0; g < 8; ++g) {
    const int qrow = g + (hiHalf ? 8 : 0);
    const float ilq = __shfl(il, qrow);
    bf16_t* op = ab + ((size_t)b * Ss + q0 + qrow) * Ee + h * Dd + qLane;
    op[0]  = (bf16_t)(o0[g] * ilq);
    op[16] = (bf16_t)(o1[g] * ilq);
    op[32] = (bf16_t)(o2[g] * ilq);
    op[48] = (bf16_t)(o3[g] * ilq);
  }
}

// ------------------------------------------------- output projection (NT GEMM)
__global__ __launch_bounds__(128) void out_gemm(const bf16_t* __restrict__ ab,
                                                const bf16_t* __restrict__ wob,
                                                float* __restrict__ out) {
  __shared__ bf16_t sB[2 * SB_TILE];
  const int wave = threadIdx.x >> 5;
  const int mblock = blockIdx.x & 31;
  const int nblock = blockIdx.x >> 5;   // 16 tiles of 64 cols
  const int m0 = mblock * 128 + wave * 32;
  const int f0 = nblock * 64;
  v8f c00 = {}, c01 = {}, c02 = {}, c03 = {};
  v8f c10 = {}, c11 = {}, c12 = {}, c13 = {};
  gemm_core_32x64(ab, wob, sB, m0, f0, c00, c01, c02, c03, c10, c11, c12, c13);

  const int lane = threadIdx.x & 31;
  const bool hiHalf = lane >= 16;
  const int qLane = lane & 15;
  const v8f cc[2][4] = {{c00, c01, c02, c03}, {c10, c11, c12, c13}};
  for (int mh = 0; mh < 2; ++mh)
    for (int nt = 0; nt < 4; ++nt)
      for (int g = 0; g < 8; ++g)
        out[(size_t)(m0 + mh * 16 + g + (hiHalf ? 8 : 0)) * Ee + f0 + nt * 16 + qLane] =
            cc[mh][nt][g];
}

// ------------------------------------------------------------------ launcher
extern "C" void kernel_launch(void* const* d_in, const int* in_sizes, int n_in,
                              void* d_out, int out_size, void* d_ws, size_t ws_size,
                              hipStream_t stream) {
  (void)in_sizes; (void)n_in; (void)out_size; (void)ws_size;
  const float* x     = (const float*)d_in[0];
  const float* w_qkv = (const float*)d_in[1];
  const float* w_out = (const float*)d_in[2];
  float* out = (float*)d_out;

  char* ws = (char*)d_ws;
  const size_t MB = 1024ull * 1024ull;
  bf16_t* xb  = (bf16_t*)(ws + 0 * MB);   // 4096x1024  (8 MB)
  bf16_t* wqb = (bf16_t*)(ws + 8 * MB);   // 3072x1024  (6 MB)
  bf16_t* wob = (bf16_t*)(ws + 14 * MB);  // 1024x1024  (2 MB)
  bf16_t* Qb  = (bf16_t*)(ws + 16 * MB);  // [2,16,2048,64] (8 MB)
  bf16_t* Kb  = (bf16_t*)(ws + 24 * MB);  // [2,16,2048,64] (8 MB)
  bf16_t* Vt  = (bf16_t*)(ws + 32 * MB);  // [2,16,64,2048] (8 MB)
  bf16_t* ab  = (bf16_t*)(ws + 40 * MB);  // 4096x1024  (8 MB)

  cvt_f32_bf16<<<2048, 256, 0, stream>>>(x,     xb,  Mm * Ee);
  cvt_f32_bf16<<<2048, 256, 0, stream>>>(w_qkv, wqb, NF * Ee);
  cvt_f32_bf16<<<1024, 256, 0, stream>>>(w_out, wob, Ee * Ee);

  // 32 M-blocks (128 rows) x 48 N-tiles (64 cols), 4 waves/block, 32x64 per wave
  qkv_gemm<<<1536, 128, 0, stream>>>(xb, wqb, Qb, Kb, Vt);
  // 2*16*128 = 4096 waves
  flash_attn<<<1024, 128, 0, stream>>>(Qb, Kb, Vt, ab);
  // 32 M-blocks x 16 N-tiles
  out_gemm<<<512, 128, 0, stream>>>(ab, wob, out);
}
